// ESNLayer_19138374271539
// MI455X (gfx1250) — matrix operand order repaired and checked by reading, not compiled, not run on hardware
//
#include <hip/hip_runtime.h>
#include <hip/hip_bf16.h>
#include <math.h>

#define T_STEPS 4096
#define NB      32      // batch
#define H       512     // hidden
#define NV      3       // input features
#define LSTR    520     // padded LDS row stride (bf16 elems); 1040 B, 16B-aligned
#define NBLOCKS 4
#define THREADS 256     // 8 waves (wave32)

typedef __attribute__((ext_vector_type(16))) __bf16 v16bf;
typedef __attribute__((ext_vector_type(8)))  float  v8f;
typedef __attribute__((ext_vector_type(4)))  unsigned int v4ui;
typedef __attribute__((ext_vector_type(8)))  int    v8i;
typedef __attribute__((ext_vector_type(4)))  int    v4i;

union Frag16 { unsigned u[8]; v16bf v; };

// f32 -> bf16 round-to-nearest-even, raw ushort
__device__ __forceinline__ unsigned short f2bf(float f) {
    unsigned x = __builtin_bit_cast(unsigned, f);
    return (unsigned short)((x + 0x7FFFu + ((x >> 16) & 1u)) >> 16);
}

// CDNA5 has a native TANH transcendental (TRANS32); fall back to exp2+rcp.
__device__ __forceinline__ float fast_tanh(float x) {
#if __has_builtin(__builtin_amdgcn_tanhf)
    return __builtin_amdgcn_tanhf(x);
#else
    float e = __builtin_amdgcn_exp2f(2.885390082f * x);   // exp(2x)
    return 1.0f - 2.0f * __builtin_amdgcn_rcpf(e + 1.0f);
#endif
}

__global__ void esn_init_kernel(unsigned* cnt) {
    if (threadIdx.x == 0) *cnt = 0u;
}

__global__ __launch_bounds__(THREADS, 1)
void esn_scan_kernel(const float* __restrict__ X,      // (NB, T, NV)
                     const float* __restrict__ Win,    // (H, NV)
                     const float* __restrict__ Wint,   // (H, H) row-major [k][n]
                     const float* __restrict__ noise,  // (T, NB, H)
                     float* __restrict__ out,          // (NB, T, H)
                     unsigned short* __restrict__ Sg,  // (NB, H) bf16 state exchange
                     unsigned* __restrict__ barrier_cnt)
{
    __shared__ unsigned short s_lds[NB * LSTR];        // 33,280 B: state s_{t-1} in bf16

    const int tid  = threadIdx.x;
    const int lane = tid & 31;          // wave32
    const int wave = tid >> 5;
    const int grp  = lane >> 4;         // half-wave
    const int ln16 = lane & 15;
    const int h    = blockIdx.x * 128 + wave * 16 + ln16;   // this lane's output column

    // Per-lane input-projection weights (column h fixed for this lane)
    const float wi0 = Win[h * NV + 0];
    const float wi1 = Win[h * NV + 1];
    const float wi2 = Win[h * NV + 2];

    // ---- TDM descriptor: load full (32 x 512) bf16 state tile Sg -> LDS,
    // with HW padding of 4 dwords after every 256 dwords => LDS row stride 520 bf16.
    const unsigned long long gaddr = (unsigned long long)(uintptr_t)Sg;
    const unsigned lds_base = (unsigned)(uintptr_t)(void*)s_lds;  // LDS byte offset
    v4ui tdm_g0;
    tdm_g0.x = 0x1u;                               // count=1 valid descriptor
    tdm_g0.y = lds_base;                           // lds_addr
    tdm_g0.z = (unsigned)(gaddr & 0xFFFFFFFFu);    // global_addr[31:0]
    tdm_g0.w = (unsigned)((gaddr >> 32) & 0x01FFFFFFu)  // global_addr[56:32]
             | 0x80000000u;                        // type=2 ("image")
    v8i tdm_g1;
    tdm_g1[0] = (1 << 16)                          // data_size = 2 bytes (bf16)
              | (1 << 20)                          // pad_enable
              | (7 << 22)                          // pad_interval: every 256 dwords (one row)
              | (3 << 25);                         // pad_amount: 4 dwords (16 B)
    tdm_g1[1] = (H & 0xFFFF) << 16;                // tensor_dim0[15:0]  = 512
    tdm_g1[2] = (H >> 16) | ((NB & 0xFFFF) << 16); // tensor_dim0[31:16], tensor_dim1 = 32
    tdm_g1[3] = (NB >> 16) | (H << 16);            // tensor_dim1 hi, tile_dim0 = 512
    tdm_g1[4] = NB;                                // tile_dim1 = 32, tile_dim2 = 0
    tdm_g1[5] = H;                                 // tensor_dim0_stride = 512 elems
    tdm_g1[6] = 0;                                 // stride hi, tensor_dim1_stride lo
    tdm_g1[7] = 0;
    v4i tdm_g2 = {0, 0, 0, 0};
    v4i tdm_g3 = {0, 0, 0, 0};
    v8i tdm_g4 = {0, 0, 0, 0, 0, 0, 0, 0};        // extra group (clang-23 6-arg form)

    // ---- Load this wave's B strip (K=512 x N=16) into registers as bf16 fragments.
    // B 32x16 bf16 packing chosen here (same k order as A): N = lane%16; K = (lane/16)*16 + r*2 + half.
    Frag16 Bf[16];
    #pragma unroll
    for (int kt = 0; kt < 16; ++kt) {
        #pragma unroll
        for (int r = 0; r < 8; ++r) {
            int k0 = kt * 32 + grp * 16 + r * 2;
            unsigned lo = f2bf(Wint[(size_t)(k0    ) * H + h]);
            unsigned hi = f2bf(Wint[(size_t)(k0 + 1) * H + h]);
            Bf[kt].u[r] = lo | (hi << 16);
        }
    }

    // ---- Initial state s_0 = 0.01 everywhere
    {
        unsigned short iv = f2bf(0.01f);
        for (int i = tid; i < NB * LSTR; i += THREADS) s_lds[i] = iv;
    }
    __syncthreads();

    const int arow = ln16;  // A-matrix row within a 16-row tile for this lane

    for (int t = 0; t < T_STEPS; ++t) {
        #pragma unroll
        for (int mt = 0; mt < 2; ++mt) {
            // ---- C init: u = 0.01*noise[t,b,h] + X[b,t,:]·Win[h,:]
            // C/D layout: M = r + grp*8 (+16*mt), N = lane%16
            v8f c;
            #pragma unroll
            for (int r = 0; r < 8; ++r) {
                int b = mt * 16 + grp * 8 + r;
                const float* xp = &X[((size_t)b * T_STEPS + t) * NV];
                float u = 0.01f * noise[((size_t)t * NB + b) * H + h]
                        + xp[0] * wi0 + xp[1] * wi1 + xp[2] * wi2;
                c[r] = u;
                if (t + 1 < T_STEPS)  // warm L2/L0 for next step's noise row
                    __builtin_prefetch(&noise[((size_t)(t + 1) * NB + b) * H + h], 0, 0);
            }

            // ---- K=512 accumulation: 16 chained bf16 WMMAs (f32 accumulate)
            // A 16x32 bf16 packing chosen here (same k order as B): M = lane%16; K = grp*16 + r*2 + half
            #pragma unroll
            for (int kt = 0; kt < 16; ++kt) {
                Frag16 a;
                #pragma unroll
                for (int r = 0; r < 8; ++r) {
                    int k0 = kt * 32 + grp * 16 + r * 2;
                    a.u[r] = *(const unsigned*)&s_lds[(mt * 16 + arow) * LSTR + k0];
                }
                c = __builtin_amdgcn_wmma_f32_16x16x32_bf16(
                        false, a.v, false, Bf[kt].v, (short)0, c, false, false);
            }

            // ---- activation + stores (f32 states out, bf16 state exchange)
            #pragma unroll
            for (int r = 0; r < 8; ++r) {
                int b = mt * 16 + grp * 8 + r;
                float s = fast_tanh(c[r]);
                out[((size_t)b * T_STEPS + t) * H + h] = s;
                Sg[(size_t)b * H + h] = f2bf(s);
            }
        }

        // ---- per-step inter-block barrier (4 blocks, monotonic counter)
        __threadfence();            // make this thread's Sg stores device-visible
        __syncthreads();            // all waves in block have fenced
        if (tid == 0)
            __hip_atomic_fetch_add(barrier_cnt, 1u, __ATOMIC_ACQ_REL,
                                   __HIP_MEMORY_SCOPE_AGENT);
        {
            const unsigned target = (unsigned)(NBLOCKS * (t + 1));
            while (__hip_atomic_load(barrier_cnt, __ATOMIC_ACQUIRE,
                                     __HIP_MEMORY_SCOPE_AGENT) < target)
                __builtin_amdgcn_s_sleep(1);
        }
        __syncthreads();

        // ---- refresh LDS state via Tensor Data Mover (async DMA, TENSORcnt)
        if (t + 1 < T_STEPS) {
            if (wave == 0) {
                __builtin_amdgcn_tensor_load_to_lds(tdm_g0, tdm_g1, tdm_g2, tdm_g3,
                                                    tdm_g4, 0);
                __builtin_amdgcn_s_wait_tensorcnt(0);
            }
        }
        __syncthreads();
    }
}

extern "C" void kernel_launch(void* const* d_in, const int* in_sizes, int n_in,
                              void* d_out, int out_size, void* d_ws, size_t ws_size,
                              hipStream_t stream) {
    const float* X     = (const float*)d_in[0];   // (32, 4096, 3)
    const float* Win   = (const float*)d_in[1];   // (512, 3)
    const float* Wint  = (const float*)d_in[2];   // (512, 512)
    const float* noise = (const float*)d_in[3];   // (4096, 32, 512)
    float* out = (float*)d_out;                   // (32, 4096, 512)

    unsigned short* Sg = (unsigned short*)d_ws;                 // 32 KB bf16 state
    unsigned* cnt      = (unsigned*)((char*)d_ws + 32 * 1024);  // barrier counter

    esn_init_kernel<<<1, 1, 0, stream>>>(cnt);
    esn_scan_kernel<<<NBLOCKS, THREADS, 0, stream>>>(X, Win, Wint, noise, out, Sg, cnt);
}
